// Block_80951543595569
// MI455X (gfx1250) — matrix operand order repaired and checked by reading, not compiled, not run on hardware
//
#include <hip/hip_runtime.h>

// ---------------- problem constants (fixed by reference) ----------------
constexpr int Bdim = 4;
constexpr int Tdim = 2048;
constexpr int Dm   = 1024;
constexpr int Hh   = 16;
constexpr int HDim = 64;
constexpr int BT   = Bdim * Tdim;     // 8192
constexpr int D4   = 4 * Dm;          // 4096
constexpr int N3   = 3 * Dm;          // 3072

// ---------------- types ----------------
typedef __bf16 bf16_t;
typedef __attribute__((ext_vector_type(16))) __bf16 bf16x16;
typedef __attribute__((ext_vector_type(8)))  float  floatx8;
typedef __attribute__((ext_vector_type(4)))  float  floatx4;

__device__ inline bf16_t to_bf16(float f) {
    union { float f; unsigned u; } in; in.f = f;
    unsigned u = in.u;
    u += 0x7fffu + ((u >> 16) & 1u);          // round-to-nearest-even
    union { unsigned short s; bf16_t b; } out;
    out.s = (unsigned short)(u >> 16);
    return out.b;
}

// A-fragment (16x32, ISA layout): lane<16 holds K {0..7,16..23}, lane>=16 holds
// K {8..15,24..31} of row M=lane&15. Caller passes p already offset by
// m*lda + koffA (koffA = 0 or 8). Two contiguous 16B runs, 16 halves apart.
__device__ inline bf16x16 loadA16(const bf16_t* p) {
    union { floatx4 f[2]; bf16x16 v; } u;
    u.f[0] = *(const floatx4*)(p);
    u.f[1] = *(const floatx4*)(p + 16);
    return u.v;
}

// B-fragment (32x16, from B^T stored [N][K]): lane<16 holds K {0..7,16..23} of column
// N=lane; lane>=16 holds K {8..15,24..31} of column N=lane-16. Caller passes p already
// offset by n*ldb + koffB (koffB = 0 or 8). Two contiguous 16B runs, 16 halves apart.
__device__ inline bf16x16 loadB16(const bf16_t* p) {
    union { floatx4 f[2]; bf16x16 v; } u;
    u.f[0] = *(const floatx4*)(p);
    u.f[1] = *(const floatx4*)(p + 16);
    return u.v;
}

__device__ inline floatx8 wmma_bf16(bf16x16 a, bf16x16 b, floatx8 c) {
    return __builtin_amdgcn_wmma_f32_16x16x32_bf16(
        false, a, false, b, (short)0, c, false, false);
}

// ---------------- weight prep ----------------
// Wq/Wk/Wv are [H][D][HD] fp32; build concatenated B^T [3D][D] bf16 with
// N index n = h*64 + kk.
__global__ __launch_bounds__(256) void k_conv_qkv(
    const float* Wq, const float* Wk, const float* Wv, bf16_t* Bt)
{
    long i = (long)blockIdx.x * 256 + threadIdx.x;
    if (i >= (long)N3 * Dm) return;
    int w  = (int)(i / ((long)Dm * Dm));
    long r = i % ((long)Dm * Dm);
    int n  = (int)(r / Dm);
    int d  = (int)(r % Dm);
    int h  = n >> 6;
    int kk = n & 63;
    const float* W = (w == 0) ? Wq : (w == 1) ? Wk : Wv;
    Bt[i] = to_bf16(W[(long)h * Dm * HDim + (long)d * HDim + kk]);
}

// W is [K][N] fp32 row-major -> Bt [N][K] bf16
__global__ __launch_bounds__(256) void k_transpose(
    const float* W, bf16_t* Bt, int K, int N)
{
    long i = (long)blockIdx.x * 256 + threadIdx.x;
    if (i >= (long)K * N) return;
    int n = (int)(i / K);
    int k = (int)(i % K);
    Bt[i] = to_bf16(W[(long)k * N + n]);
}

// ---------------- layernorm: fp32 [rows][D] -> bf16 [rows][D] ----------------
__global__ __launch_bounds__(256) void k_ln(
    const float* __restrict__ x, const float* __restrict__ g,
    const float* __restrict__ be, bf16_t* __restrict__ out)
{
    int row = blockIdx.x;
    int tid = threadIdx.x;
    const float* xr = x + (long)row * Dm;
    floatx4 raw = ((const floatx4*)xr)[tid];
    float v[4] = { raw[0], raw[1], raw[2], raw[3] };
    float s  = v[0] + v[1] + v[2] + v[3];
    float ss = v[0]*v[0] + v[1]*v[1] + v[2]*v[2] + v[3]*v[3];
    __shared__ float r1[256];
    __shared__ float r2[256];
    r1[tid] = s; r2[tid] = ss;
    __syncthreads();
    for (int st = 128; st > 0; st >>= 1) {
        if (tid < st) { r1[tid] += r1[tid + st]; r2[tid] += r2[tid + st]; }
        __syncthreads();
    }
    float mu   = r1[0] * (1.0f / Dm);
    float var  = r2[0] * (1.0f / Dm) - mu * mu;
    float rstd = rsqrtf(var + 1e-5f);
    int base = tid * 4;
    #pragma unroll
    for (int j = 0; j < 4; j++) {
        out[(long)row * Dm + base + j] =
            to_bf16((v[j] - mu) * rstd * g[base + j] + be[base + j]);
    }
}

// ---------------- generic WMMA GEMM (ping-pong pipelined, 32x64/wave) --------
// C[M][N] = A[M][K](bf16) x Bt[N][K](bf16). Each wave owns a 32x64 tile:
// 2 A-fragments x 4 B-fragments -> 8 WMMA per K=32 step. K-loop unrolled by 2
// into a true ping-pong: each buffer is loaded exactly once per iteration
// (after the WMMAs that consume its previous value), so no register copies
// and no VALU->WMMA-source hazards. KDIM compile-time => all fragment strides
// are instruction-immediate offsets.
// MODE 0: QKV routing (N=3D): q,k row-major [B][H][T][64]; v transposed [B][H][64][T]
// MODE 1: outF = resid + acc + bias (f32)
// MODE 2: outB = bf16(relu(acc + bias))
template <int KDIM, int MODE>
__global__ __launch_bounds__(256) void k_gemm(
    const bf16_t* __restrict__ A, const bf16_t* __restrict__ Bt,
    int M, int N,
    const float* __restrict__ bias, const float* __restrict__ resid,
    float* __restrict__ outF, bf16_t* __restrict__ outB,
    bf16_t* __restrict__ qb, bf16_t* __restrict__ kb, bf16_t* __restrict__ vT)
{
    static_assert((KDIM / 32) % 2 == 0, "K/32 must be even for ping-pong");
    int lane = threadIdx.x & 31;
    int wave = threadIdx.x >> 5;
    long tile = (long)blockIdx.x * 8 + wave;
    int mt_n = M >> 5, nt_n = N >> 6;
    if (tile >= (long)mt_n * nt_n) return;
    int mt = (int)(tile % mt_n);   // consecutive waves share nt -> B hits in L0
    int nt = (int)(tile / mt_n);

    int m     = lane & 15;
    int hi    = lane >> 4;
    int koffA = hi << 3;    // 0 or 8
    int koffB = hi << 3;    // 0 or 8

    floatx8 acc[2][4];
    #pragma unroll
    for (int i = 0; i < 2; i++)
        #pragma unroll
        for (int j = 0; j < 4; j++) acc[i][j] = {};

    const bf16_t* Arow = A + (long)(mt * 32 + m) * KDIM + koffA;
    const bf16_t* Bcol = Bt + (long)(nt * 64 + m) * KDIM + koffB;

    bf16x16 a0[2], b0[4], a1[2], b1[4];
    #pragma unroll
    for (int i = 0; i < 2; i++) {
        a0[i] = loadA16(Arow + i * 16 * KDIM);
        a1[i] = loadA16(Arow + i * 16 * KDIM + 32);
    }
    #pragma unroll
    for (int j = 0; j < 4; j++) {
        b0[j] = loadB16(Bcol + j * 16 * KDIM);
        b1[j] = loadB16(Bcol + j * 16 * KDIM + 32);
    }

    for (int k0 = 0; k0 < KDIM - 64; k0 += 64) {
        // step k0: consume buf0, then refill it for k0+64
        #pragma unroll
        for (int i = 0; i < 2; i++)
            #pragma unroll
            for (int j = 0; j < 4; j++)
                acc[i][j] = wmma_bf16(a0[i], b0[j], acc[i][j]);
        #pragma unroll
        for (int i = 0; i < 2; i++) a0[i] = loadA16(Arow + i * 16 * KDIM + k0 + 64);
        #pragma unroll
        for (int j = 0; j < 4; j++) b0[j] = loadB16(Bcol + j * 16 * KDIM + k0 + 64);
        __builtin_prefetch(Arow + k0 + 256, 0, 3);   // global_prefetch_b8

        // step k0+32: consume buf1, then refill it for k0+96
        #pragma unroll
        for (int i = 0; i < 2; i++)
            #pragma unroll
            for (int j = 0; j < 4; j++)
                acc[i][j] = wmma_bf16(a1[i], b1[j], acc[i][j]);
        #pragma unroll
        for (int i = 0; i < 2; i++) a1[i] = loadA16(Arow + i * 16 * KDIM + k0 + 96);
        #pragma unroll
        for (int j = 0; j < 4; j++) b1[j] = loadB16(Bcol + j * 16 * KDIM + k0 + 96);
    }
    // tail: K-64 and K-32
    #pragma unroll
    for (int i = 0; i < 2; i++)
        #pragma unroll
        for (int j = 0; j < 4; j++)
            acc[i][j] = wmma_bf16(a0[i], b0[j], acc[i][j]);
    #pragma unroll
    for (int i = 0; i < 2; i++)
        #pragma unroll
        for (int j = 0; j < 4; j++)
            acc[i][j] = wmma_bf16(a1[i], b1[j], acc[i][j]);

    // epilogue: acc[i][j][r] -> row = mt*32 + i*16 + hi*8 + r, col = nt*64 + j*16 + m
    #pragma unroll
    for (int i = 0; i < 2; i++) {
        int rowbase = mt * 32 + i * 16 + hi * 8;
        #pragma unroll
        for (int j = 0; j < 4; j++) {
            int c = nt * 64 + j * 16 + m;
            #pragma unroll
            for (int r = 0; r < 8; r++) {
                int row = rowbase + r;
                float val = acc[i][j][r];
                if (MODE == 0) {
                    int b = row >> 11;          // row / T
                    int t = row & (Tdim - 1);
                    if (c < Dm) {
                        int h = c >> 6, kk = c & 63;
                        qb[(((long)b * Hh + h) * Tdim + t) * HDim + kk] = to_bf16(val);
                    } else if (c < 2 * Dm) {
                        int c2 = c - Dm;
                        int h = c2 >> 6, kk = c2 & 63;
                        kb[(((long)b * Hh + h) * Tdim + t) * HDim + kk] = to_bf16(val);
                    } else {
                        int c2 = c - 2 * Dm;
                        int h = c2 >> 6, kk = c2 & 63;
                        vT[(((long)b * Hh + h) * HDim + kk) * Tdim + t] = to_bf16(val);
                    }
                } else if (MODE == 1) {
                    long idx = (long)row * N + c;
                    outF[idx] = resid[idx] + val + bias[c];
                } else { // MODE 2
                    long idx = (long)row * N + c;
                    outB[idx] = to_bf16(fmaxf(val + bias[c], 0.0f));
                }
            }
        }
    }
}

// ---------------- causal flash attention (pipelined) ----------------
// One wave per (b,h, 16-row query tile). q,k: [B][H][T][64] bf16; vT: [B][H][64][T].
// Output Ob: [B][T][D] bf16 (heads concatenated).
// Row-sums of the probability tile are computed with an extra WMMA against an
// all-ones B operand (P x 1 -> row-sum broadcast across the 16 lanes of each
// half-wave in C-layout), replacing 32 ds_bpermute ops per chunk.
__global__ __launch_bounds__(32) void k_attn(
    const bf16_t* __restrict__ q, const bf16_t* __restrict__ k,
    const bf16_t* __restrict__ vT, bf16_t* __restrict__ Ob)
{
    int t0   = blockIdx.x * 16;
    int bh   = blockIdx.y;
    int b    = bh >> 4;
    int h    = bh & 15;
    int lane = threadIdx.x;
    int m    = lane & 15;
    int hi   = lane >> 4;
    int koffA = hi << 3;
    int koffB = hi << 3;

    const bf16_t* qbase = q + ((long)bh * Tdim + t0) * HDim + (long)m * HDim + koffA;
    bf16x16 qf0 = loadA16(qbase);        // K 0..31 slice
    bf16x16 qf1 = loadA16(qbase + 32);   // K 32..63 slice

    // all-ones B fragment (bf16 1.0 = 0x3F80)
    union { unsigned u[8]; bf16x16 v; } onesu;
    #pragma unroll
    for (int i = 0; i < 8; i++) onesu.u[i] = 0x3F803F80u;
    const bf16x16 onesB = onesu.v;

    float rmax[8], rsum[8], corr[8];
    floatx8 oacc[4];
    #pragma unroll
    for (int r = 0; r < 8; r++) { rmax[r] = -3.0e38f; rsum[r] = 0.0f; }
    #pragma unroll
    for (int j = 0; j < 4; j++) oacc[j] = {};

    __shared__ __align__(16) bf16_t pl[16 * 32];

    const bf16_t* kroot = k  + (long)bh * Tdim * HDim;
    const bf16_t* vb    = vT + (long)bh * HDim * Tdim;

    auto loadK = [&](int c0, bf16x16* f) {
        const bf16_t* kbp = kroot + (long)(c0 + m) * HDim + koffB;
        f[0] = loadB16(kbp);                  // cols c0..c0+15, K 0..31
        f[1] = loadB16(kbp + 32);             // cols c0..c0+15, K 32..63
        f[2] = loadB16(kbp + 16 * HDim);      // cols c0+16..c0+31, K 0..31
        f[3] = loadB16(kbp + 16 * HDim + 32); // cols c0+16..c0+31, K 32..63
    };

    bf16x16 kc[4];
    loadK(0, kc);
    int cEnd = t0 + 16;

    for (int c0 = 0; c0 < cEnd; c0 += 32) {
        floatx8 s0 = {};
        floatx8 s1 = {};
        s0 = wmma_bf16(qf0, kc[0], s0);
        s0 = wmma_bf16(qf1, kc[1], s0);
        s1 = wmma_bf16(qf0, kc[2], s1);
        s1 = wmma_bf16(qf1, kc[3], s1);

        // refill kc in place for the next chunk (after the WMMAs consumed it),
        // and issue this chunk's V loads; the softmax VALU burst hides latency
        if (c0 + 32 < cEnd) loadK(c0 + 32, kc);
        bf16x16 vc[4];
        #pragma unroll
        for (int j = 0; j < 4; j++)
            vc[j] = loadB16(vb + (long)(j * 16 + m) * Tdim + c0 + koffB);

        int ca = c0 + m;
        int cb = c0 + 16 + m;
        #pragma unroll
        for (int r = 0; r < 8; r++) {
            int t = t0 + hi * 8 + r;
            float sa = (ca <= t) ? s0[r] * 0.125f : -3.0e38f;
            float sb = (cb <= t) ? s1[r] * 0.125f : -3.0e38f;
            float v = fmaxf(sa, sb);
            #pragma unroll
            for (int off = 1; off < 16; off <<= 1)
                v = fmaxf(v, __shfl_xor(v, off, 32));
            float mnew = fmaxf(rmax[r], v);
            corr[r] = __expf(rmax[r] - mnew);
            rmax[r] = mnew;
            float pa = __expf(sa - mnew);
            float pb = __expf(sb - mnew);
            #pragma unroll
            for (int j = 0; j < 4; j++) oacc[j][r] *= corr[r];
            // stash probabilities as a 16x32 bf16 tile (row = matrix row M)
            int prow = hi * 8 + r;
            pl[prow * 32 + m]      = to_bf16(pa);
            pl[prow * 32 + 16 + m] = to_bf16(pb);
        }
        asm volatile("s_wait_dscnt 0" ::: "memory");

        bf16x16 pfrag = loadA16(pl + (long)m * 32 + koffA);
        floatx8 sums = {};
        sums = wmma_bf16(pfrag, onesB, sums);   // row-sum broadcast per C-layout
        #pragma unroll
        for (int j = 0; j < 4; j++)
            oacc[j] = wmma_bf16(pfrag, vc[j], oacc[j]);
        #pragma unroll
        for (int r = 0; r < 8; r++)
            rsum[r] = rsum[r] * corr[r] + sums[r];
    }

    #pragma unroll
    for (int j = 0; j < 4; j++) {
        #pragma unroll
        for (int r = 0; r < 8; r++) {
            int t = t0 + hi * 8 + r;
            float val = oacc[j][r] / rsum[r];
            Ob[((long)b * Tdim + t) * Dm + h * HDim + j * 16 + m] = to_bf16(val);
        }
    }
}

// ---------------- host orchestration ----------------
extern "C" void kernel_launch(void* const* d_in, const int* in_sizes, int n_in,
                              void* d_out, int out_size, void* d_ws, size_t ws_size,
                              hipStream_t stream)
{
    (void)in_sizes; (void)n_in; (void)out_size; (void)ws_size;
    const float* x   = (const float*)d_in[0];
    const float* Wq  = (const float*)d_in[1];
    const float* Wk  = (const float*)d_in[2];
    const float* Wv  = (const float*)d_in[3];
    const float* Wp  = (const float*)d_in[4];
    const float* bp  = (const float*)d_in[5];
    const float* W1  = (const float*)d_in[6];
    const float* b1  = (const float*)d_in[7];
    const float* W2  = (const float*)d_in[8];
    const float* b2  = (const float*)d_in[9];
    const float* g1  = (const float*)d_in[10];
    const float* be1 = (const float*)d_in[11];
    const float* g2  = (const float*)d_in[12];
    const float* be2 = (const float*)d_in[13];
    float* out = (float*)d_out;

    // workspace carve-up
    char* ws = (char*)d_ws;
    size_t o = 0;
    bf16_t* hb     = (bf16_t*)(ws + o); o += (size_t)BT * Dm * 2;     // LN1 out
    bf16_t* qbuf   = (bf16_t*)(ws + o); o += (size_t)BT * Dm * 2;
    bf16_t* kbuf   = (bf16_t*)(ws + o); o += (size_t)BT * Dm * 2;
    bf16_t* vTbuf  = (bf16_t*)(ws + o); o += (size_t)BT * Dm * 2;
    bf16_t* Ob     = (bf16_t*)(ws + o); o += (size_t)BT * Dm * 2;     // attn out (concat heads)
    float*  x2     = (float*) (ws + o); o += (size_t)BT * Dm * 4;     // residual stream after attn
    bf16_t* h2b    = (bf16_t*)(ws + o); o += (size_t)BT * Dm * 2;     // LN2 out
    bf16_t* ffb    = (bf16_t*)(ws + o); o += (size_t)BT * D4 * 2;     // relu(ffn1)
    bf16_t* WqkvT  = (bf16_t*)(ws + o); o += (size_t)N3 * Dm * 2;
    bf16_t* WpT    = (bf16_t*)(ws + o); o += (size_t)Dm * Dm * 2;
    bf16_t* W1T    = (bf16_t*)(ws + o); o += (size_t)D4 * Dm * 2;
    bf16_t* W2T    = (bf16_t*)(ws + o); o += (size_t)Dm * D4 * 2;

    // 1) weight prep
    k_conv_qkv<<<(N3 * Dm) / 256, 256, 0, stream>>>(Wq, Wk, Wv, WqkvT);
    k_transpose<<<(Dm * Dm) / 256, 256, 0, stream>>>(Wp, WpT, Dm, Dm);
    k_transpose<<<(Dm * D4) / 256, 256, 0, stream>>>(W1, W1T, Dm, D4);
    k_transpose<<<(D4 * Dm) / 256, 256, 0, stream>>>(W2, W2T, D4, Dm);

    // 2) LN1
    k_ln<<<BT, 256, 0, stream>>>(x, g1, be1, hb);

    // 3) fused QKV projection (M=BT, N=3D, K=D), routed epilogue
    {
        int tiles = (BT / 32) * (N3 / 64);
        k_gemm<Dm, 0><<<tiles / 8, 256, 0, stream>>>(hb, WqkvT, BT, N3,
            nullptr, nullptr, nullptr, nullptr, qbuf, kbuf, vTbuf);
    }

    // 4) causal flash attention
    k_attn<<<dim3(Tdim / 16, Bdim * Hh), 32, 0, stream>>>(qbuf, kbuf, vTbuf, Ob);

    // 5) output projection + residual: x2 = x + Ob*Wp + bp
    {
        int tiles = (BT / 32) * (Dm / 64);
        k_gemm<Dm, 1><<<tiles / 8, 256, 0, stream>>>(Ob, WpT, BT, Dm,
            bp, x, x2, nullptr, nullptr, nullptr, nullptr);
    }

    // 6) LN2
    k_ln<<<BT, 256, 0, stream>>>(x2, g2, be2, h2b);

    // 7) FFN up + ReLU: ffb = relu(h2b*W1 + b1)
    {
        int tiles = (BT / 32) * (D4 / 64);
        k_gemm<Dm, 2><<<tiles / 8, 256, 0, stream>>>(h2b, W1T, BT, D4,
            b1, nullptr, nullptr, ffb, nullptr, nullptr, nullptr);
    }

    // 8) FFN down + residual: out = x2 + ffb*W2 + b2
    {
        int tiles = (BT / 32) * (Dm / 64);
        k_gemm<D4, 1><<<tiles / 8, 256, 0, stream>>>(ffb, W2T, BT, Dm,
            b2, x2, out, nullptr, nullptr, nullptr, nullptr);
    }
}
